// GraphomerBlock_67345087201718
// MI455X (gfx1250) — matrix operand order repaired and checked by reading, not compiled, not run on hardware
//
#include <hip/hip_runtime.h>
#include <hip/hip_bf16.h>

typedef _Float16 f16;
typedef _Float16 v8h  __attribute__((ext_vector_type(8)));
typedef _Float16 v16h __attribute__((ext_vector_type(16)));
typedef float    v8f  __attribute__((ext_vector_type(8)));

#define D_MODEL 768
#define NHEADS  8
#define DH      96
#define D_FF    3072
#define BATCH   8
#define SEQ     1024
#define ROWS    (BATCH * SEQ)   // 8192

static __device__ inline v16h cat16(v8h lo, v8h hi) {
    return __builtin_shufflevector(lo, hi, 0,1,2,3,4,5,6,7,8,9,10,11,12,13,14,15);
}

static __device__ inline v8f wmma_f16(v16h a, v16h b, v8f c) {
    // D = A(16x32 f16) * B(32x16 f16) + C(16x16 f32)
    return __builtin_amdgcn_wmma_f32_16x16x32_f16(false, a, false, b, (short)0, c, false, false);
}

// A-fragment (16x32, f16): lane m = l&15; kh = l>>4.
// elems 0..7 -> K = kh*8 + i ; elems 8..15 -> K = 16 + kh*8 + i
static __device__ inline v16h load_afrag(const f16* rowptr, int kh) {
    const f16* p = rowptr + kh * 8;
    v8h lo = *(const v8h*)p;
    v8h hi = *(const v8h*)(p + 16);
    return cat16(lo, hi);
}

// B-fragment (32x16, f16) from K-contiguous storage (BT row = column n):
// lane n = l&15; kh = l>>4; elems 0..7 -> K = kh*8 + i ; elems 8..15 -> K = 16 + kh*8 + i
static __device__ inline v16h load_bfrag(const f16* rowptr, int kh) {
    const f16* p = rowptr + kh * 8;
    v8h lo = *(const v8h*)p;
    v8h hi = *(const v8h*)(p + 16);
    return cat16(lo, hi);
}

// ---------------------------------------------------------------------------
// CDNA5 async global->LDS copy (16 bytes per lane), tracked by ASYNCcnt.
// LDS byte address = low 32 bits of the flat LDS pointer (ISA 10.2: the LDS
// aperture keeps the LDS offset in addr[31:0]).
// ---------------------------------------------------------------------------
static __device__ inline void async_ld16(void* lds, const void* gptr) {
    unsigned l = (unsigned)(unsigned long long)lds;
    asm volatile("global_load_async_to_lds_b128 %0, %1, off"
                 : : "v"(l), "v"(gptr) : "memory");
}
static __device__ inline void wait_async() {
    asm volatile("s_wait_asynccnt 0x0" : : : "memory");
}

// ---------------------------------------------------------------------------
// f32 -> f16 elementwise convert
// ---------------------------------------------------------------------------
__global__ __launch_bounds__(256)
void k_cvt_f16(const float* __restrict__ in, f16* __restrict__ out, long n) {
    for (long i = (long)blockIdx.x * 256 + threadIdx.x; i < n; i += 256L * gridDim.x)
        out[i] = (f16)in[i];
}

// ---------------------------------------------------------------------------
// W[K][N] f32 -> WT[N][K] f16  (so GEMM B tiles are K-contiguous)
// ---------------------------------------------------------------------------
__global__ __launch_bounds__(256)
void k_transpose_cvt(const float* __restrict__ in, f16* __restrict__ out, int K, int N) {
    long total = (long)K * N;
    for (long i = (long)blockIdx.x * 256 + threadIdx.x; i < total; i += 256L * gridDim.x) {
        int k = (int)(i / N);
        int nn = (int)(i % N);
        out[(long)nn * K + k] = (f16)in[i];
    }
}

// ---------------------------------------------------------------------------
// GEMM: C[M,N] = A[M,K](f16) * BT[N,K](f16)^T + bias[N]; optional SiLU.
// Block tile 128x128, BK=32, double-buffered LDS, async global->LDS staging.
// 8 waves, each owning a 32x64 sub-tile = 8 WMMA accumulators.
// ---------------------------------------------------------------------------
__global__ __launch_bounds__(256)
void k_gemm_f16(const f16* __restrict__ A, const f16* __restrict__ BT,
                const float* __restrict__ bias,
                float* __restrict__ Cf, f16* __restrict__ Ch,
                int M, int N, int K, int act_silu)
{
    __shared__ f16 As[2][128][48];   // 32 + 16 pad (rows stay 16B-aligned)
    __shared__ f16 Bs[2][128][48];

    const int tid  = threadIdx.x;
    const int lane = tid & 31;
    const int wave = tid >> 5;
    const int tm0  = blockIdx.x * 128;
    const int tn0  = blockIdx.y * 128;
    const int wm   = (wave & 3) * 32;
    const int wn   = (wave >> 2) * 64;
    const int lm   = lane & 15;
    const int kh   = lane >> 4;

    const int r4 = tid >> 2;   // 0..63
    const int s4 = tid & 3;    // 0..3

    v8f acc[2][4] = {};
    const int ntile = K >> 5;

    auto stage = [&](int buf, int k0) {
        async_ld16(&As[buf][r4][s4 * 8],      &A[(long)(tm0 + r4)      * K + k0 + s4 * 8]);
        async_ld16(&As[buf][r4 + 64][s4 * 8], &A[(long)(tm0 + r4 + 64) * K + k0 + s4 * 8]);
        async_ld16(&Bs[buf][r4][s4 * 8],      &BT[(long)(tn0 + r4)      * K + k0 + s4 * 8]);
        async_ld16(&Bs[buf][r4 + 64][s4 * 8], &BT[(long)(tn0 + r4 + 64) * K + k0 + s4 * 8]);
    };

    stage(0, 0);
    for (int t = 0; t < ntile; ++t) {
        const int p = t & 1;
        wait_async();
        __syncthreads();
        if (t + 1 < ntile) stage(p ^ 1, (t + 1) << 5);   // DMA next tile under compute

        v16h af[2], bf[4];
#pragma unroll
        for (int mf = 0; mf < 2; ++mf)
            af[mf] = load_afrag(&As[p][wm + mf * 16 + lm][0], kh);
#pragma unroll
        for (int nf = 0; nf < 4; ++nf)
            bf[nf] = load_bfrag(&Bs[p][wn + nf * 16 + lm][0], kh);

#pragma unroll
        for (int mf = 0; mf < 2; ++mf)
#pragma unroll
            for (int nf = 0; nf < 4; ++nf)
                acc[mf][nf] = wmma_f16(af[mf], bf[nf], acc[mf][nf]);
    }

    // epilogue: C/D layout -> row = base + r + 8*kh, col = base + lm
#pragma unroll
    for (int mf = 0; mf < 2; ++mf) {
#pragma unroll
        for (int nf = 0; nf < 4; ++nf) {
#pragma unroll
            for (int r = 0; r < 8; ++r) {
                int row = tm0 + wm + mf * 16 + r + 8 * kh;
                int col = tn0 + wn + nf * 16 + lm;
                float v = acc[mf][nf][r] + bias[col];
                if (act_silu) v = v / (1.0f + __expf(-v));
                long o = (long)row * N + col;
                if (Cf) Cf[o] = v;
                if (Ch) Ch[o] = (f16)v;
            }
        }
    }
}

// ---------------------------------------------------------------------------
// Flash attention: ctx[b,n,h*96+d] = softmax(Q K^T * s + bias) V
// Block = (b, h, 128 q rows); wave = 16 q rows; KV in double-buffered
// 32-key tiles: K staged by async DMA, V prefetched to regs + transposed.
// ---------------------------------------------------------------------------
__global__ __launch_bounds__(256)
void k_attn(const f16* __restrict__ Q, const f16* __restrict__ Kh,
            const f16* __restrict__ Vh, const float* __restrict__ bias,
            f16* __restrict__ ctx)
{
    __shared__ f16 Ks[2][32][112];    // [key][d]  (96 + 16 pad)
    __shared__ f16 Vt[2][96][48];     // [d][key]  (32 + 16 pad)
    __shared__ f16 Pw[8][16][48];     // per-wave P relayout scratch

    const int tid  = threadIdx.x;
    const int lane = tid & 31;
    const int wave = tid >> 5;
    const int bid  = blockIdx.x;
    const int b    = bid >> 6;          // 8 q-tiles * 8 heads per batch
    const int h    = (bid >> 3) & 7;
    const int q0   = (bid & 7) * 128 + wave * 16;
    const int lm   = lane & 15;
    const int kh   = lane >> 4;
    const long bS  = (long)b * SEQ;
    const int hoff = h * DH;
    const float scale = 0.10206207261596577f;  // 1/sqrt(96)

    // Q fragments for this wave's 16 rows: 3 K-chunks of 32 over dh=96
    v16h aQ[3];
    {
        const f16* qrow = Q + (bS + q0 + lm) * D_MODEL + hoff;
#pragma unroll
        for (int kc = 0; kc < 3; ++kc)
            aQ[kc] = load_afrag(qrow + kc * 32, kh);
    }

    // staging work split for K tile: 384 x 16B segments
    auto stageK = [&](int buf, int kv0) {
        int r = tid / 12, s = tid % 12;
        async_ld16(&Ks[buf][r][s * 8], &Kh[(bS + kv0 + r) * D_MODEL + hoff + s * 8]);
        if (tid < 128) {
            int idx2 = tid + 256;
            int r2 = idx2 / 12, s2 = idx2 % 12;
            async_ld16(&Ks[buf][r2][s2 * 8], &Kh[(bS + kv0 + r2) * D_MODEL + hoff + s2 * 8]);
        }
    };

    f16 vv[12];
    auto loadV = [&](int kv0) {
#pragma unroll
        for (int i = 0; i < 12; ++i) {
            int idx = tid + i * 256;       // 0..3071
            int key = idx / 96, d = idx % 96;
            vv[i] = Vh[(bS + kv0 + key) * D_MODEL + hoff + d];
        }
    };
    auto storeV = [&](int buf) {
#pragma unroll
        for (int i = 0; i < 12; ++i) {
            int idx = tid + i * 256;
            int key = idx / 96, d = idx % 96;
            Vt[buf][d][key] = vv[i];
        }
    };

    float m_i[8], l_i[8];
    v8f O[6] = {};
#pragma unroll
    for (int r = 0; r < 8; ++r) { m_i[r] = -3.0e38f; l_i[r] = 0.0f; }

    // prologue: tile 0 into buffer 0
    stageK(0, 0);
    loadV(0);
    storeV(0);

    const int NT = SEQ / 32;   // 32 tiles
    for (int t = 0; t < NT; ++t) {
        const int p   = t & 1;
        const int kv0 = t * 32;
        wait_async();
        __syncthreads();                         // Ks[p], Vt[p] visible to all
        if (t + 1 < NT) { stageK(p ^ 1, kv0 + 32); loadV(kv0 + 32); }

        // S = Q K^T : two 16x16 accumulators cover 32 keys
        v8f S0 = {}, S1 = {};
#pragma unroll
        for (int kc = 0; kc < 3; ++kc) {
            v16h b0 = load_bfrag(&Ks[p][lm][kc * 32], kh);
            v16h b1 = load_bfrag(&Ks[p][16 + lm][kc * 32], kh);
            S0 = wmma_f16(aQ[kc], b0, S0);
            S1 = wmma_f16(aQ[kc], b1, S1);
        }

        // scale + bias + online softmax (row = q0 + r + 8*kh)
        const float* bb = bias + bS * SEQ;
#pragma unroll
        for (int r = 0; r < 8; ++r) {
            int row = q0 + r + 8 * kh;
            float s0 = S0[r] * scale + bb[(long)row * SEQ + kv0 + lm];
            float s1 = S1[r] * scale + bb[(long)row * SEQ + kv0 + 16 + lm];
            float mx = fmaxf(s0, s1);
#pragma unroll
            for (int off = 1; off < 16; off <<= 1)
                mx = fmaxf(mx, __shfl_xor(mx, off, 32));
            float newm = fmaxf(m_i[r], mx);
            float fac  = __expf(m_i[r] - newm);
            m_i[r] = newm;
            float p0 = __expf(s0 - newm);
            float p1 = __expf(s1 - newm);
            float rs = p0 + p1;
#pragma unroll
            for (int off = 1; off < 16; off <<= 1)
                rs += __shfl_xor(rs, off, 32);
            l_i[r] = l_i[r] * fac + rs;
#pragma unroll
            for (int dt = 0; dt < 6; ++dt)
                O[dt][r] *= fac;
            int ml = r + 8 * kh;
            Pw[wave][ml][lm]      = (f16)p0;
            Pw[wave][ml][16 + lm] = (f16)p1;
        }
        asm volatile("s_wait_dscnt 0x0" ::: "memory");

        // O += P(16x32) @ V(32x96) : 6 WMMAs over d-tiles
        v16h aP = load_afrag(&Pw[wave][lm][0], kh);
#pragma unroll
        for (int dt = 0; dt < 6; ++dt) {
            v16h bV = load_bfrag(&Vt[p][dt * 16 + lm][0], kh);
            O[dt] = wmma_f16(aP, bV, O[dt]);
        }

        if (t + 1 < NT) storeV(p ^ 1);   // transpose-store next V under the loop
    }

    // finalize and write ctx (f16, feeds the O-projection GEMM)
#pragma unroll
    for (int dt = 0; dt < 6; ++dt) {
#pragma unroll
        for (int r = 0; r < 8; ++r) {
            int row = q0 + r + 8 * kh;
            int d   = dt * 16 + lm;
            ctx[(bS + row) * D_MODEL + hoff + d] = (f16)(O[dt][r] / l_i[r]);
        }
    }
}

// ---------------------------------------------------------------------------
// y = alpha * rmsnorm(xa + xb) + gamma ; one wave per 768-element row
// ---------------------------------------------------------------------------
__global__ __launch_bounds__(256)
void k_add_rmsnorm(const float* __restrict__ xa, const float* __restrict__ xb,
                   const float* __restrict__ alpha, const float* __restrict__ gamma,
                   float* __restrict__ outf, f16* __restrict__ outh)
{
    const int row  = blockIdx.x * 8 + (threadIdx.x >> 5);
    const int lane = threadIdx.x & 31;
    const long base = (long)row * D_MODEL;

    float v[24];
    float ss = 0.0f;
#pragma unroll
    for (int i = 0; i < 24; ++i) {
        int c = lane + i * 32;
        float t = xa[base + c] + xb[base + c];
        v[i] = t;
        ss += t * t;
    }
#pragma unroll
    for (int off = 16; off >= 1; off >>= 1)
        ss += __shfl_xor(ss, off, 32);
    float rr = rsqrtf(ss * (1.0f / D_MODEL) + 1e-6f);
#pragma unroll
    for (int i = 0; i < 24; ++i) {
        int c = lane + i * 32;
        float o = alpha[c] * (v[i] * rr) + gamma[c];
        outf[base + c] = o;
        if (outh) outh[base + c] = (f16)o;
    }
}

// ---------------------------------------------------------------------------
extern "C" void kernel_launch(void* const* d_in, const int* in_sizes, int n_in,
                              void* d_out, int out_size, void* d_ws, size_t ws_size,
                              hipStream_t stream) {
    (void)in_sizes; (void)n_in; (void)out_size; (void)ws_size;

    const float* x         = (const float*)d_in[0];
    const float* attn_bias = (const float*)d_in[1];
    const float* wq = (const float*)d_in[2];  const float* bq = (const float*)d_in[3];
    const float* wk = (const float*)d_in[4];  const float* bk = (const float*)d_in[5];
    const float* wv = (const float*)d_in[6];  const float* bv = (const float*)d_in[7];
    const float* wo = (const float*)d_in[8];  const float* bo = (const float*)d_in[9];
    const float* w1 = (const float*)d_in[10]; const float* b1 = (const float*)d_in[11];
    const float* w2 = (const float*)d_in[12]; const float* b2 = (const float*)d_in[13];
    const float* alpha1 = (const float*)d_in[14]; const float* gamma1 = (const float*)d_in[15];
    const float* alpha2 = (const float*)d_in[16]; const float* gamma2 = (const float*)d_in[17];

    char* ws = (char*)d_ws;
    size_t off = 0;
    auto alloc = [&](size_t bytes) -> void* {
        void* p = ws + off;
        off += (bytes + 255) & ~(size_t)255;
        return p;
    };

    f16*   x_h   = (f16*)alloc((size_t)ROWS * D_MODEL * 2);
    f16*   wqT   = (f16*)alloc((size_t)D_MODEL * D_MODEL * 2);
    f16*   wkT   = (f16*)alloc((size_t)D_MODEL * D_MODEL * 2);
    f16*   wvT   = (f16*)alloc((size_t)D_MODEL * D_MODEL * 2);
    f16*   woT   = (f16*)alloc((size_t)D_MODEL * D_MODEL * 2);
    f16*   w1T   = (f16*)alloc((size_t)D_MODEL * D_FF * 2);   // [3072][768]
    f16*   w2T   = (f16*)alloc((size_t)D_FF * D_MODEL * 2);   // [768][3072]
    f16*   q_h   = (f16*)alloc((size_t)ROWS * D_MODEL * 2);
    f16*   k_h   = (f16*)alloc((size_t)ROWS * D_MODEL * 2);
    f16*   v_h   = (f16*)alloc((size_t)ROWS * D_MODEL * 2);
    f16*   ctx_h = (f16*)alloc((size_t)ROWS * D_MODEL * 2);
    float* aout  = (float*)alloc((size_t)ROWS * D_MODEL * 4);
    float* x1    = (float*)alloc((size_t)ROWS * D_MODEL * 4);
    f16*   x1_h  = (f16*)alloc((size_t)ROWS * D_MODEL * 2);
    f16*   h1    = (f16*)alloc((size_t)ROWS * D_FF * 2);
    float* ff    = (float*)alloc((size_t)ROWS * D_MODEL * 4);

    dim3 blk(256);

    // precision prep: activations + transposed weights in f16
    k_cvt_f16<<<4096, blk, 0, stream>>>(x, x_h, (long)ROWS * D_MODEL);
    k_transpose_cvt<<<1152, blk, 0, stream>>>(wq, wqT, D_MODEL, D_MODEL);
    k_transpose_cvt<<<1152, blk, 0, stream>>>(wk, wkT, D_MODEL, D_MODEL);
    k_transpose_cvt<<<1152, blk, 0, stream>>>(wv, wvT, D_MODEL, D_MODEL);
    k_transpose_cvt<<<1152, blk, 0, stream>>>(wo, woT, D_MODEL, D_MODEL);
    k_transpose_cvt<<<2048, blk, 0, stream>>>(w1, w1T, D_MODEL, D_FF);
    k_transpose_cvt<<<2048, blk, 0, stream>>>(w2, w2T, D_FF, D_MODEL);

    // QKV projections
    dim3 gdd(ROWS / 128, D_MODEL / 128);   // 64 x 6
    k_gemm_f16<<<gdd, blk, 0, stream>>>(x_h, wqT, bq, nullptr, q_h, ROWS, D_MODEL, D_MODEL, 0);
    k_gemm_f16<<<gdd, blk, 0, stream>>>(x_h, wkT, bk, nullptr, k_h, ROWS, D_MODEL, D_MODEL, 0);
    k_gemm_f16<<<gdd, blk, 0, stream>>>(x_h, wvT, bv, nullptr, v_h, ROWS, D_MODEL, D_MODEL, 0);

    // attention with edge bias
    k_attn<<<BATCH * NHEADS * (SEQ / 128), blk, 0, stream>>>(q_h, k_h, v_h, attn_bias, ctx_h);

    // output projection
    k_gemm_f16<<<gdd, blk, 0, stream>>>(ctx_h, woT, bo, aout, nullptr, ROWS, D_MODEL, D_MODEL, 0);

    // residual + RMSNorm 1 (f32 for next residual, f16 for FFN GEMM)
    k_add_rmsnorm<<<ROWS / 8, blk, 0, stream>>>(x, aout, alpha1, gamma1, x1, x1_h);

    // FFN
    dim3 gdf(ROWS / 128, D_FF / 128);      // 64 x 24
    k_gemm_f16<<<gdf, blk, 0, stream>>>(x1_h, w1T, b1, nullptr, h1, ROWS, D_FF, D_MODEL, 1);
    k_gemm_f16<<<gdd, blk, 0, stream>>>(h1, w2T, b2, ff, nullptr, ROWS, D_MODEL, D_FF, 0);

    // residual + RMSNorm 2 -> final output
    k_add_rmsnorm<<<ROWS / 8, blk, 0, stream>>>(x1, ff, alpha2, gamma2, (float*)d_out, nullptr);
}
